// CrossAttention_45612552683928
// MI455X (gfx1250) — matrix operand order repaired and checked by reading, not compiled, not run on hardware
//
#include <hip/hip_runtime.h>
#include <hip/hip_bf16.h>
#include <math.h>

// ---------------------------------------------------------------------------
// CrossAttention (RMSNorm -> Q/K proj -> mean-over-heads softmax -> attn @ V)
// B=2, Nq=4096, Nk=2048, D=128, H=4, hd=32, Dv=768.
// f16 WMMA (v_wmma_f32_16x16x32_f16) for QK^T and P@V.
// Pass 1: raw-max sweep (no muls) + sumexp sweep (fma + native exp2 / elem).
// Pass 2: per-head probabilities = one fma + one v_exp per element (scale,
// max-shift and 0.25/l all folded into the exp2 argument), LDS head-sum,
// double-buffered LDS (1 barrier/chunk), branchless prefetch.
// ---------------------------------------------------------------------------

#define B_   2
#define NQ   4096
#define NK   2048
#define DIM  128
#define NH   4
#define HD   32
#define DV   768
#define SCALE 0.17677669529663687f            // 1/sqrt(32)
#define LOG2E 1.4426950408889634f
#define S2    (SCALE * LOG2E)                 // folds score scale into exp2 arg
#define RMS_EPS 1.1920929e-07f

typedef __attribute__((ext_vector_type(16))) _Float16 v16h;
typedef __attribute__((ext_vector_type(8)))  float    v8f;

static __device__ __forceinline__ v8f wmma_f16(v16h a, v16h b, v8f c) {
  return __builtin_amdgcn_wmma_f32_16x16x32_f16(false, a, false, b,
                                                (short)0, c, false, false);
}

static __device__ __forceinline__ v16h load_b16(const _Float16* p) {
  v16h v;
  #pragma unroll
  for (int i = 0; i < 16; ++i) v[i] = p[(i & 7) + ((i >> 3) << 4)];
  return v;
}

// ---------------------------------------------------------------------------
// Kernel 1: fused RMSNorm + linear projection (x @ W^T + b), output f16.
// ---------------------------------------------------------------------------
__global__ void rmsnorm_proj_kernel(const float* __restrict__ x,
                                    const float* __restrict__ wrms,
                                    const float* __restrict__ W,
                                    const float* __restrict__ bias,
                                    _Float16* __restrict__ y) {
  __shared__ float xs[DIM];
  __shared__ float red[DIM];
  const int row = blockIdx.x;
  const int tid = threadIdx.x;
  float xv = x[(size_t)row * DIM + tid];
  red[tid] = xv * xv;
  __syncthreads();
  #pragma unroll
  for (int s = 64; s > 0; s >>= 1) {
    if (tid < s) red[tid] += red[tid + s];
    __syncthreads();
  }
  const float inv = rsqrtf(red[0] * (1.0f / DIM) + RMS_EPS);
  xs[tid] = xv * inv * wrms[tid];
  __syncthreads();
  const float* wr = W + (size_t)tid * DIM;
  float acc = bias[tid];
  #pragma unroll 8
  for (int d = 0; d < DIM; ++d) acc = fmaf(xs[d], wr[d], acc);
  y[(size_t)row * DIM + tid] = (_Float16)acc;
}

// ---------------------------------------------------------------------------
// Kernel 2: V [B][Nk][Dv] f32  ->  Vt [B][Dv][Nk] f16 (tiled transpose).
// ---------------------------------------------------------------------------
__global__ void v_transpose_half_kernel(const float* __restrict__ v,
                                        _Float16* __restrict__ vt) {
  __shared__ _Float16 tile[32][33];
  const int b  = blockIdx.z;
  const int k0 = blockIdx.x * 32;
  const int d0 = blockIdx.y * 32;
  const int tx = threadIdx.x;
  const int ty = threadIdx.y;
  #pragma unroll
  for (int i = ty; i < 32; i += 8)
    tile[i][tx] = (_Float16)v[((size_t)b * NK + (k0 + i)) * DV + (d0 + tx)];
  __syncthreads();
  #pragma unroll
  for (int i = ty; i < 32; i += 8)
    vt[((size_t)b * DV + (d0 + i)) * NK + (k0 + tx)] = tile[tx][i];
}

// ---------------------------------------------------------------------------
// Kernel 3 (pass 1): per-(b,h,qrow) m = max_k(s), l = sum_k exp(s - m).
// Sweep A: raw max (scale deferred).  Sweep B: l += exp2(fma(d, S2, -m*S2)).
// ---------------------------------------------------------------------------
__global__ void attn_stats_kernel(const _Float16* __restrict__ qh,
                                  const _Float16* __restrict__ kh,
                                  float* __restrict__ m_out,
                                  float* __restrict__ l_out) {
  const int tileId = blockIdx.x;                // 0 .. B*NQ/16-1
  const int b      = tileId / (NQ / 16);
  const int qbase  = (tileId % (NQ / 16)) * 16;
  const int h      = threadIdx.x >> 5;          // wave = head
  const int lane   = threadIdx.x & 31;
  const int seg    = lane >> 4;
  const int l16    = lane & 15;

  const _Float16* abase =
      qh + ((size_t)b * NQ + qbase + l16) * DIM + h * HD + seg * 8;
  v16h aq;
  #pragma unroll
  for (int i = 0; i < 8; ++i) { aq[i] = abase[i]; aq[8 + i] = abase[16 + i]; }

  const _Float16* kbase0 =
      kh + ((size_t)b * NK + l16) * DIM + h * HD + seg * 8;

  // ---- Sweep A: raw row max (scale applied once at the end) ----
  float mr[8];
  #pragma unroll
  for (int r = 0; r < 8; ++r) mr[r] = -__builtin_inff();
  {
    v16h bk_cur = load_b16(kbase0);
    for (int kt = 0; kt < NK / 16; ++kt) {
      const int ktn = (kt + 1) & (NK / 16 - 1);   // branchless wrap
      v16h bk_nxt = load_b16(kbase0 + (size_t)ktn * 16 * DIM);
      v8f d;
      #pragma unroll
      for (int r = 0; r < 8; ++r) d[r] = 0.0f;
      d = wmma_f16(aq, bk_cur, d);
      #pragma unroll
      for (int r = 0; r < 8; ++r) mr[r] = fmaxf(mr[r], d[r]);
      bk_cur = bk_nxt;
    }
  }
  #pragma unroll
  for (int r = 0; r < 8; ++r) {
    #pragma unroll
    for (int mask = 1; mask < 16; mask <<= 1)
      mr[r] = fmaxf(mr[r], __shfl_xor(mr[r], mask, 32));
  }

  // ---- Sweep B: sumexp, one fma + one native exp2 per element ----
  float lr[8], me2[8];
  #pragma unroll
  for (int r = 0; r < 8; ++r) { lr[r] = 0.0f; me2[r] = mr[r] * S2; }
  {
    v16h bk_cur = load_b16(kbase0);
    for (int kt = 0; kt < NK / 16; ++kt) {
      const int ktn = (kt + 1) & (NK / 16 - 1);
      v16h bk_nxt = load_b16(kbase0 + (size_t)ktn * 16 * DIM);
      v8f d;
      #pragma unroll
      for (int r = 0; r < 8; ++r) d[r] = 0.0f;
      d = wmma_f16(aq, bk_cur, d);
      #pragma unroll
      for (int r = 0; r < 8; ++r)
        lr[r] += __builtin_amdgcn_exp2f(fmaf(d[r], S2, -me2[r]));
      bk_cur = bk_nxt;
    }
  }
  #pragma unroll
  for (int r = 0; r < 8; ++r) {
    #pragma unroll
    for (int mask = 1; mask < 16; mask <<= 1)
      lr[r] += __shfl_xor(lr[r], mask, 32);
  }

  if (l16 == 0) {
    #pragma unroll
    for (int r = 0; r < 8; ++r) {
      const int row  = qbase + r + seg * 8;
      const size_t i = (size_t)(b * NH + h) * NQ + row;
      m_out[i] = mr[r] * SCALE;     // scaled max (external semantics)
      l_out[i] = lr[r];
    }
  }
}

// ---------------------------------------------------------------------------
// Kernel 4 (pass 2): scores for own head -> p = exp2(fma(d, S2, c_r)) with
// c_r = log2(0.25/l_r) - m_r*log2e  (scale, shift, 1/(4l) all folded) ->
// LDS -> head-sum into A layout -> 12 PV WMMAs per wave (192-dim slice).
// ---------------------------------------------------------------------------
__global__ void attn_pv_kernel(const _Float16* __restrict__ qh,
                               const _Float16* __restrict__ kh,
                               const _Float16* __restrict__ vt,
                               const float* __restrict__ m_in,
                               const float* __restrict__ l_in,
                               float* __restrict__ out) {
  __shared__ float pS[2][NH][16][33];           // per-head probabilities
  const int tileId = blockIdx.x;
  const int b      = tileId / (NQ / 16);
  const int qbase  = (tileId % (NQ / 16)) * 16;
  const int w      = threadIdx.x >> 5;          // head for scores, dim slice for PV
  const int lane   = threadIdx.x & 31;
  const int seg    = lane >> 4;
  const int l16    = lane & 15;

  const _Float16* abase =
      qh + ((size_t)b * NQ + qbase + l16) * DIM + w * HD + seg * 8;
  v16h aq;
  #pragma unroll
  for (int i = 0; i < 8; ++i) { aq[i] = abase[i]; aq[8 + i] = abase[16 + i]; }

  // Per-lane exp2 constants for this wave's head, rows M = seg*8 + r.
  float cr[8];
  #pragma unroll
  for (int r = 0; r < 8; ++r) {
    const size_t i = (size_t)(b * NH + w) * NQ + qbase + seg * 8 + r;
    cr[r] = __builtin_amdgcn_logf(0.25f / l_in[i]) - m_in[i] * LOG2E;
  }

  v8f acc[12];
  #pragma unroll
  for (int t = 0; t < 12; ++t)
    #pragma unroll
    for (int r = 0; r < 8; ++r) acc[t][r] = 0.0f;

  const _Float16* kbase =
      kh + ((size_t)b * NK + l16) * DIM + w * HD + seg * 8;
  const _Float16* vbase =
      vt + ((size_t)b * DV + w * 192 + l16) * NK + seg * 8;

  v16h bk0 = load_b16(kbase);                       // chunk 0, tile 0
  v16h bk1 = load_b16(kbase + (size_t)16 * DIM);    // chunk 0, tile 1

  for (int kc = 0; kc < NK / 32; ++kc) {
    const int par = kc & 1;
    // ---- scores for own head (2 tiles of 16 keys) ----
    v8f d0, d1;
    #pragma unroll
    for (int r = 0; r < 8; ++r) { d0[r] = 0.0f; d1[r] = 0.0f; }
    d0 = wmma_f16(aq, bk0, d0);
    d1 = wmma_f16(aq, bk1, d1);
    // prefetch next chunk's key tiles (branchless wrap, hidden under PV)
    const int kcn = (kc + 1) & (NK / 32 - 1);
    bk0 = load_b16(kbase + (size_t)(kcn * 32) * DIM);
    bk1 = load_b16(kbase + (size_t)(kcn * 32 + 16) * DIM);
    // ---- probabilities: one fma + one v_exp per element ----
    #pragma unroll
    for (int r = 0; r < 8; ++r) {
      pS[par][w][seg * 8 + r][l16] =
          __builtin_amdgcn_exp2f(fmaf(d0[r], S2, cr[r]));
      pS[par][w][seg * 8 + r][16 + l16] =
          __builtin_amdgcn_exp2f(fmaf(d1[r], S2, cr[r]));
    }
    __syncthreads();
    // ---- head-sum at read side, directly in A layout ----
    v16h p;
    #pragma unroll
    for (int i = 0; i < 8; ++i) {
      const int c0 = seg * 8 + i;
      const int c1 = 16 + seg * 8 + i;
      const float s0 = pS[par][0][l16][c0] + pS[par][1][l16][c0]
                     + pS[par][2][l16][c0] + pS[par][3][l16][c0];
      const float s1 = pS[par][0][l16][c1] + pS[par][1][l16][c1]
                     + pS[par][2][l16][c1] + pS[par][3][l16][c1];
      p[i]     = (_Float16)s0;
      p[8 + i] = (_Float16)s1;
    }
    // ---- P @ V over this wave's 192-dim slice, one-deep V prefetch ----
    const _Float16* vkc = vbase + (size_t)kc * 32;
    v16h bv_cur = load_b16(vkc);
    #pragma unroll
    for (int t = 0; t < 12; ++t) {
      const int tn = (t + 1 < 12) ? t + 1 : t;
      v16h bv_nxt = load_b16(vkc + (size_t)(tn * 16) * NK);
      acc[t] = wmma_f16(p, bv_cur, acc[t]);
      bv_cur = bv_nxt;
    }
  }

  #pragma unroll
  for (int t = 0; t < 12; ++t) {
    const int dimc = w * 192 + t * 16 + l16;
    #pragma unroll
    for (int r = 0; r < 8; ++r) {
      const int row = qbase + r + seg * 8;
      out[((size_t)b * NQ + row) * DV + dimc] = acc[t][r];
    }
  }
}

// ---------------------------------------------------------------------------
// Launch
// ---------------------------------------------------------------------------
extern "C" void kernel_launch(void* const* d_in, const int* in_sizes, int n_in,
                              void* d_out, int out_size, void* d_ws, size_t ws_size,
                              hipStream_t stream) {
  const float* query = (const float*)d_in[0];   // [B,NQ,128]
  const float* key   = (const float*)d_in[1];   // [B,NK,128]
  const float* value = (const float*)d_in[2];   // [B,NK,768]
  const float* rmsq  = (const float*)d_in[3];
  const float* rmsk  = (const float*)d_in[4];
  const float* Wq    = (const float*)d_in[5];
  const float* Wk    = (const float*)d_in[6];
  const float* bq    = (const float*)d_in[7];
  const float* bk    = (const float*)d_in[8];
  float* out = (float*)d_out;

  // Workspace layout (~9.7 MB total)
  char* ws = (char*)d_ws;
  _Float16* qh = (_Float16*)(ws);                                  // 2 MB
  _Float16* kh = (_Float16*)(ws + (size_t)B_ * NQ * DIM * 2);      // 1 MB
  _Float16* vt = (_Float16*)(ws + (size_t)B_ * (NQ + NK) * DIM * 2);   // 6 MB
  float*    ms = (float*)(ws + (size_t)B_ * (NQ + NK) * DIM * 2
                             + (size_t)B_ * DV * NK * 2);          // 256 KB stats
  float*    ls = ms + (size_t)B_ * NH * NQ;

  rmsnorm_proj_kernel<<<B_ * NQ, DIM, 0, stream>>>(query, rmsq, Wq, bq, qh);
  rmsnorm_proj_kernel<<<B_ * NK, DIM, 0, stream>>>(key,   rmsk, Wk, bk, kh);
  v_transpose_half_kernel<<<dim3(NK / 32, DV / 32, B_), dim3(32, 8), 0, stream>>>(value, vt);
  attn_stats_kernel<<<B_ * NQ / 16, 128, 0, stream>>>(qh, kh, ms, ls);
  attn_pv_kernel<<<B_ * NQ / 16, 128, 0, stream>>>(qh, kh, vt, ms, ls, out);
}
